// EmbeddingPredictor_25056839206008
// MI455X (gfx1250) — matrix operand order repaired and checked by reading, not compiled, not run on hardware
//
#include <hip/hip_runtime.h>
#include <math.h>

typedef float v8f  __attribute__((ext_vector_type(8)));
typedef __bf16 v16bf __attribute__((ext_vector_type(16)));
typedef __bf16 v8bf  __attribute__((ext_vector_type(8)));

#define B_ 16
#define T_ 2048
#define E_ 512
#define H_ 4
#define C_ 3
#define M_ (B_ * T_)          // 32768 tokens
#define EPS 1e-5f

// ----- fp32 <-> bf16 helpers (RNE), bit-level to avoid header type friction -----
__device__ __forceinline__ unsigned short f2bf(float f) {
    unsigned u = __float_as_uint(f);
    u += 0x7FFFu + ((u >> 16) & 1u);          // round-to-nearest-even
    return (unsigned short)(u >> 16);
}
__device__ __forceinline__ float bf2f(unsigned short h) {
    return __uint_as_float(((unsigned)h) << 16);
}

// ---------------------------------------------------------------------------
// Kernel 0: possum[c][e] = sum_h pos[h][e][c]   (head dimension collapses)
// ---------------------------------------------------------------------------
__global__ void k_possum(const float* __restrict__ pos, float* __restrict__ possum) {
    int idx = blockIdx.x * 256 + threadIdx.x;
    if (idx < C_ * E_) {
        int c = idx / E_, e = idx % E_;
        float s = 0.f;
        #pragma unroll
        for (int h = 0; h < H_; ++h) s += pos[h * E_ * C_ + e * C_ + c];
        possum[c * E_ + e] = s;
    }
}

// ---------------------------------------------------------------------------
// Kernel W: split ffn_w into bf16 hi/lo (W = Wh + Wl to ~16 mantissa bits)
// ---------------------------------------------------------------------------
__global__ void k_wsplit(const float* __restrict__ W,
                         unsigned short* __restrict__ Wh,
                         unsigned short* __restrict__ Wl) {
    int idx = blockIdx.x * 256 + threadIdx.x;   // E_*E_ = 262144 exact
    float x = W[idx];
    unsigned short h = f2bf(x);
    Wh[idx] = h;
    Wl[idx] = f2bf(x - bf2f(h));
}

// ---------------------------------------------------------------------------
// Kernel A: fused embedding gather + positional mixing; emits split-bf16 X.
// One wave per token; lane owns 16 consecutive e's. All math in fp32.
// ---------------------------------------------------------------------------
__global__ __launch_bounds__(256)
void k_embed_mix(const int* __restrict__ tokens,
                 const float* __restrict__ embed,
                 const float* __restrict__ possum,
                 unsigned short* __restrict__ Xh,
                 unsigned short* __restrict__ Xl) {
    const int wave = threadIdx.x >> 5;
    const int lane = threadIdx.x & 31;
    const int m = blockIdx.x * 8 + wave;
    const int b = m / T_;
    const int t = m % T_;
    const int e0 = lane * 16;

    float xw[C_][16];
    float s[C_];

    #pragma unroll
    for (int c = 0; c < C_; ++c) {
        int tt = t + c - (C_ - 1);
        if (tt >= 0) {
            int tok = tokens[b * T_ + tt];
            const float4* row = reinterpret_cast<const float4*>(embed + (size_t)tok * E_ + e0);
            #pragma unroll
            for (int i = 0; i < 4; ++i) {
                float4 v = row[i];
                xw[c][i*4+0] = v.x; xw[c][i*4+1] = v.y;
                xw[c][i*4+2] = v.z; xw[c][i*4+3] = v.w;
            }
        } else {
            #pragma unroll
            for (int i = 0; i < 16; ++i) xw[c][i] = 0.f;
        }
        const float4* ps = reinterpret_cast<const float4*>(possum + c * E_ + e0);
        float d = 0.f;
        #pragma unroll
        for (int i = 0; i < 4; ++i) {
            float4 p = ps[i];
            d += xw[c][i*4+0]*p.x + xw[c][i*4+1]*p.y + xw[c][i*4+2]*p.z + xw[c][i*4+3]*p.w;
        }
        #pragma unroll
        for (int off = 1; off < 32; off <<= 1) d += __shfl_xor(d, off, 32);
        s[c] = d;
    }

    const float inv = 1.0f / (float)(H_ * C_);
    float o[16];
    #pragma unroll
    for (int i = 0; i < 16; ++i)
        o[i] = (s[0]*xw[0][i] + s[1]*xw[1][i] + s[2]*xw[2][i]) * inv;

    // pack hi/lo bf16, 2 per dword, 8 dwords per lane (32B contiguous)
    uint4 ph[2], pl[2];
    #pragma unroll
    for (int q = 0; q < 2; ++q) {
        unsigned hh[4], ll[4];
        #pragma unroll
        for (int j = 0; j < 4; ++j) {
            float x0 = o[q*8 + j*2], x1 = o[q*8 + j*2 + 1];
            unsigned short h0 = f2bf(x0), h1 = f2bf(x1);
            unsigned short l0 = f2bf(x0 - bf2f(h0)), l1 = f2bf(x1 - bf2f(h1));
            hh[j] = (unsigned)h0 | ((unsigned)h1 << 16);
            ll[j] = (unsigned)l0 | ((unsigned)l1 << 16);
        }
        ph[q] = make_uint4(hh[0], hh[1], hh[2], hh[3]);
        pl[q] = make_uint4(ll[0], ll[1], ll[2], ll[3]);
    }
    uint4* xhp = reinterpret_cast<uint4*>(Xh + (size_t)m * E_ + e0);
    uint4* xlp = reinterpret_cast<uint4*>(Xl + (size_t)m * E_ + e0);
    xhp[0] = ph[0]; xhp[1] = ph[1];
    xlp[0] = pl[0]; xlp[1] = pl[1];
}

// ---------------------------------------------------------------------------
// Kernel B: Y = X @ W^T + b via split-bf16 WMMA (3 products), fused LN+swish.
// WG = 512 threads (16 waves) -> tile 64(M) x 512(N): full LN row in one WG.
// Waves 2(M) x 8(N); each wave owns 32x64 = 2x4 tiles of V_WMMA_F32_16X16X32_BF16.
// Fragments come straight from global (row-major bf16) in contiguous k order (same for A and B):
//   A 16x32: lane<16 holds row m, K=[0..16); lane>=16 holds row m, K=[16..32)
//   B 32x16: lane<16 holds col n, K=[0..16); lane>=16 holds col n, K=[16..32)
// ---------------------------------------------------------------------------
__global__ __launch_bounds__(512)
void k_gemm_ln(const unsigned short* __restrict__ Xh, const unsigned short* __restrict__ Xl,
               const unsigned short* __restrict__ Wh, const unsigned short* __restrict__ Wl,
               const float* __restrict__ bias, const float* __restrict__ g,
               const float* __restrict__ bb, float* __restrict__ out) {
    __shared__ float sBias[E_], sG[E_], sBb[E_];
    __shared__ float rSum[64], rSumSq[64];

    const int tid  = threadIdx.x;
    const int wave = tid >> 5;
    const int lane = tid & 31;
    const int l    = lane & 15;
    const int half = lane >> 4;
    const int wm   = wave & 1;        // M band: 0..1 (32 rows)
    const int wn   = wave >> 1;       // N band: 0..7 (64 cols)
    const int rowBase = blockIdx.x * 64;

    sBias[tid] = bias[tid];
    sG[tid]    = g[tid];
    sBb[tid]   = bb[tid];
    if (tid < 64) { rSum[tid] = 0.f; rSumSq[tid] = 0.f; }
    __syncthreads();

    const char* xhB = (const char*)Xh;
    const char* xlB = (const char*)Xl;
    const char* whB = (const char*)Wh;
    const char* wlB = (const char*)Wl;

    v8f acc[2][4];
    #pragma unroll
    for (int i = 0; i < 2; ++i)
        #pragma unroll
        for (int j = 0; j < 4; ++j)
            #pragma unroll
            for (int r = 0; r < 8; ++r) acc[i][j][r] = 0.f;

    // per-lane row/col bases (bytes); row stride = E_*2 = 1024 B
    size_t aOff[2];
    #pragma unroll
    for (int mt = 0; mt < 2; ++mt)
        aOff[mt] = (size_t)(rowBase + wm*32 + mt*16 + l) * (E_*2) + 32*half;
    size_t bOff[4];
    #pragma unroll
    for (int nt = 0; nt < 4; ++nt)
        bOff[nt] = (size_t)(wn*64 + nt*16 + l) * (E_*2) + 32*half;

    for (int kc = 0; kc < E_; kc += 32) {
        const size_t kB = (size_t)kc * 2;

        v16bf Ah[2], Al[2];
        #pragma unroll
        for (int mt = 0; mt < 2; ++mt) {
            v8bf h1 = *reinterpret_cast<const v8bf*>(xhB + aOff[mt] + kB);
            v8bf h2 = *reinterpret_cast<const v8bf*>(xhB + aOff[mt] + kB + 16);
            Ah[mt] = __builtin_shufflevector(h1, h2, 0,1,2,3,4,5,6,7,8,9,10,11,12,13,14,15);
            v8bf l1 = *reinterpret_cast<const v8bf*>(xlB + aOff[mt] + kB);
            v8bf l2 = *reinterpret_cast<const v8bf*>(xlB + aOff[mt] + kB + 16);
            Al[mt] = __builtin_shufflevector(l1, l2, 0,1,2,3,4,5,6,7,8,9,10,11,12,13,14,15);
        }
        v16bf Bh[4], Bl[4];
        #pragma unroll
        for (int nt = 0; nt < 4; ++nt) {
            Bh[nt] = *reinterpret_cast<const v16bf*>(whB + bOff[nt] + kB);
            Bl[nt] = *reinterpret_cast<const v16bf*>(wlB + bOff[nt] + kB);
        }
        if (kc + 32 < E_) {   // prefetch next X chunk (global_prefetch_b8)
            __builtin_prefetch(xhB + aOff[0] + kB + 64, 0, 0);
            __builtin_prefetch(xlB + aOff[0] + kB + 64, 0, 0);
        }

        #pragma unroll
        for (int mt = 0; mt < 2; ++mt)
            #pragma unroll
            for (int nt = 0; nt < 4; ++nt) {
                acc[mt][nt] = __builtin_amdgcn_wmma_f32_16x16x32_bf16(
                    false, Ah[mt], false, Bh[nt], (short)0, acc[mt][nt], false, false);
                acc[mt][nt] = __builtin_amdgcn_wmma_f32_16x16x32_bf16(
                    false, Ah[mt], false, Bl[nt], (short)0, acc[mt][nt], false, false);
                acc[mt][nt] = __builtin_amdgcn_wmma_f32_16x16x32_bf16(
                    false, Al[mt], false, Bh[nt], (short)0, acc[mt][nt], false, false);
            }
    }

    // ---- bias + per-row stats (C/D: VGPR r -> row r / r+8, col = lane) ----
    #pragma unroll
    for (int mt = 0; mt < 2; ++mt) {
        #pragma unroll
        for (int nt = 0; nt < 4; ++nt) {
            const int col = wn * 64 + nt * 16 + l;
            #pragma unroll
            for (int r = 0; r < 8; ++r) {
                float v = acc[mt][nt][r] + sBias[col];
                acc[mt][nt][r] = v;
                float vs = v, vq = v * v;
                #pragma unroll
                for (int off = 1; off < 16; off <<= 1) {   // reduce each 16-lane half
                    vs += __shfl_xor(vs, off, 32);
                    vq += __shfl_xor(vq, off, 32);
                }
                if (l == 0) {
                    int row = wm * 32 + mt * 16 + r + 8 * half;
                    atomicAdd(&rSum[row], vs);
                    atomicAdd(&rSumSq[row], vq);
                }
            }
        }
    }
    __syncthreads();
    if (tid < 64) {
        float mu  = rSum[tid] * (1.0f / E_);
        float var = rSumSq[tid] * (1.0f / E_) - mu * mu;
        rSum[tid]   = mu;
        rSumSq[tid] = rsqrtf(var + EPS);
    }
    __syncthreads();

    #pragma unroll
    for (int mt = 0; mt < 2; ++mt) {
        #pragma unroll
        for (int nt = 0; nt < 4; ++nt) {
            const int col = wn * 64 + nt * 16 + l;
            #pragma unroll
            for (int r = 0; r < 8; ++r) {
                const int row = wm * 32 + mt * 16 + r + 8 * half;
                float y = (acc[mt][nt][r] - rSum[row]) * rSumSq[row] * sG[col] + sBb[col];
                y = y / (1.0f + __expf(-y));   // swish
                out[(size_t)(rowBase + row) * E_ + col] = y;
            }
        }
    }
}

// ---------------------------------------------------------------------------
extern "C" void kernel_launch(void* const* d_in, const int* in_sizes, int n_in,
                              void* d_out, int out_size, void* d_ws, size_t ws_size,
                              hipStream_t stream) {
    const int*   tokens = (const int*)  d_in[0];
    const float* embed  = (const float*)d_in[1];
    const float* pos    = (const float*)d_in[2];
    const float* ffn_w  = (const float*)d_in[3];
    const float* ffn_b  = (const float*)d_in[4];
    const float* ln_g   = (const float*)d_in[5];
    const float* ln_b   = (const float*)d_in[6];
    float* out = (float*)d_out;

    // workspace layout (bytes):
    //   [0,      16K)  possum (C*E fp32)
    //   [16K,   528K)  Wh   (E*E bf16)
    //   [528K, 1040K)  Wl   (E*E bf16)
    //   [2M,    34M )  Xh   (M*E bf16)
    //   [34M,   66M )  Xl   (M*E bf16)
    char* ws = (char*)d_ws;
    float*          possum = (float*)ws;
    unsigned short* Wh = (unsigned short*)(ws + (16u << 10));
    unsigned short* Wl = (unsigned short*)(ws + (528u << 10));
    unsigned short* Xh = (unsigned short*)(ws + ((size_t)2 << 20));
    unsigned short* Xl = (unsigned short*)(ws + ((size_t)34 << 20));

    k_possum<<<(C_ * E_ + 255) / 256, 256, 0, stream>>>(pos, possum);
    k_wsplit<<<(E_ * E_) / 256, 256, 0, stream>>>(ffn_w, Wh, Wl);
    k_embed_mix<<<M_ / 8, 256, 0, stream>>>(tokens, embed, possum, Xh, Xl);
    k_gemm_ln<<<M_ / 64, 512, 0, stream>>>(Xh, Xl, Wh, Wl, ffn_b, ln_g, ln_b, out);
}
